// ProjectionHead_5119601016941
// MI455X (gfx1250) — matrix operand order repaired and checked by reading, not compiled, not run on hardware
//
#include <hip/hip_runtime.h>

// ---------------------------------------------------------------------------
// ProjectionHead: BatchNorm1d(train) -> ReLU -> Linear
//   x:[32768,4096] f32, gamma/beta:[4096], W:[256,4096], b:[256]
//   out:[32768,256] f32
// Strategy: fp32 stats, fold BN into (scale,shift), convert activations to
// bf16 on the fly and use v_wmma_f32_16x16x32_bf16 for the projection.
// ---------------------------------------------------------------------------

typedef __attribute__((ext_vector_type(16))) __bf16 v16bf;
typedef __attribute__((ext_vector_type(8)))  float  v8f;

#define N_ROWS 32768
#define D_DIM  4096
#define O_DIM  256
#define KSTEP  32

union BFrag { v16bf v; uint4 q[2]; };

__device__ __forceinline__ unsigned short f32_to_bf16_bits(float f) {
    unsigned u = __float_as_uint(f);
    u += 0x7FFFu + ((u >> 16) & 1u);          // round-to-nearest-even
    return (unsigned short)(u >> 16);
}

// ---------------------------------------------------------------------------
// Pass 1: per-column partial sums / sums-of-squares (float4 coalesced).
// grid = (4 column-groups of 1024, 128 row-chunks of 256), block = 256.
// ---------------------------------------------------------------------------
__global__ __launch_bounds__(256)
void bn_stats_kernel(const float* __restrict__ x,
                     float* __restrict__ sum, float* __restrict__ sumsq) {
    const int c4 = blockIdx.x * 1024 + threadIdx.x * 4;
    const int r0 = blockIdx.y * 256;
    float s0 = 0.f, s1 = 0.f, s2 = 0.f, s3 = 0.f;
    float q0 = 0.f, q1 = 0.f, q2 = 0.f, q3 = 0.f;
    #pragma unroll 4
    for (int r = 0; r < 256; ++r) {
        const float4 v = *(const float4*)(x + (size_t)(r0 + r) * D_DIM + c4);
        s0 += v.x; s1 += v.y; s2 += v.z; s3 += v.w;
        q0 = fmaf(v.x, v.x, q0); q1 = fmaf(v.y, v.y, q1);
        q2 = fmaf(v.z, v.z, q2); q3 = fmaf(v.w, v.w, q3);
    }
    atomicAdd(&sum[c4 + 0], s0);  atomicAdd(&sum[c4 + 1], s1);
    atomicAdd(&sum[c4 + 2], s2);  atomicAdd(&sum[c4 + 3], s3);
    atomicAdd(&sumsq[c4 + 0], q0); atomicAdd(&sumsq[c4 + 1], q1);
    atomicAdd(&sumsq[c4 + 2], q2); atomicAdd(&sumsq[c4 + 3], q3);
}

// ---------------------------------------------------------------------------
// Pass 2: fold BN into per-feature scale/shift:
//   scale = gamma * rsqrt(var+eps),  shift = beta - mean*scale
// ---------------------------------------------------------------------------
__global__ __launch_bounds__(256)
void bn_finalize_kernel(const float* __restrict__ sum, const float* __restrict__ sumsq,
                        const float* __restrict__ gamma, const float* __restrict__ beta,
                        float* __restrict__ scale, float* __restrict__ shift) {
    const int i = blockIdx.x * 256 + threadIdx.x;
    const float invN = 1.0f / (float)N_ROWS;
    const float mean = sum[i] * invN;
    const float var  = fmaf(-mean, mean, sumsq[i] * invN);   // biased variance
    const float sc   = gamma[i] * rsqrtf(var + 1e-5f);
    scale[i] = sc;
    shift[i] = fmaf(-mean, sc, beta[i]);
}

// ---------------------------------------------------------------------------
// Pass 3: convert W [256,4096] f32 -> bf16 (lives in 192MB L2 thereafter).
// ---------------------------------------------------------------------------
__global__ __launch_bounds__(256)
void wconv_kernel(const float* __restrict__ W, unsigned short* __restrict__ Wb) {
    const int i4 = (blockIdx.x * 256 + threadIdx.x) * 4;
    const float4 v = *(const float4*)(W + i4);
    unsigned lo = (unsigned)f32_to_bf16_bits(v.x) | ((unsigned)f32_to_bf16_bits(v.y) << 16);
    unsigned hi = (unsigned)f32_to_bf16_bits(v.z) | ((unsigned)f32_to_bf16_bits(v.w) << 16);
    uint2 p; p.x = lo; p.y = hi;
    *(uint2*)(Wb + i4) = p;
}

// ---------------------------------------------------------------------------
// Pass 4: fused normalize+ReLU+GEMM with bf16 WMMA.
// Block = 256 thr (8 waves). Block covers 128 rows x all 256 cols.
// Wave w: rows [blk*128 + w*16, +16), 16 col-tiles of 16 -> 16 v8f accums.
// K-loop: 32 per step; A frag built from fp32 x via LDS-resident scale/shift.
// ---------------------------------------------------------------------------
__global__ __launch_bounds__(256)
void fused_bn_gemm_kernel(const float* __restrict__ x,
                          const float* __restrict__ scale,
                          const float* __restrict__ shift,
                          const unsigned short* __restrict__ Wb,
                          const float* __restrict__ bias,
                          float* __restrict__ out) {
    __shared__ float s_scale[D_DIM];
    __shared__ float s_shift[D_DIM];
    for (int i = threadIdx.x; i < D_DIM; i += 256) {
        s_scale[i] = scale[i];
        s_shift[i] = shift[i];
    }
    __syncthreads();

    const int lane = threadIdx.x & 31;
    const int wave = threadIdx.x >> 5;
    const int half = lane >> 4;      // lane group (0: lanes 0-15, 1: lanes 16-31)
    const int mn   = lane & 15;      // A-row / B-col within tile

    const int rowA = blockIdx.x * 128 + wave * 16 + mn;   // A matrix row for this lane
    const float* xrow = x + (size_t)rowA * D_DIM;

    v8f acc[16] = {};                // 16 col-tiles, f32 C/D fragments

    #pragma unroll 2
    for (int k0 = 0; k0 < D_DIM; k0 += KSTEP) {
        // ---- A fragment: 16x32 bf16, ISA wave32 layout ----
        // lanes 0-15: elems 0-7 -> K=k0+0..7,   elems 8-15 -> K=k0+16..23
        // lanes16-31: elems 0-7 -> K=k0+8..15,  elems 8-15 -> K=k0+24..31
        const int kA = k0 + half * 8;
        const int kB = kA + 16;

        float av[16], scv[16], shv[16];
        *(float4*)(av + 0)  = *(const float4*)(xrow + kA);
        *(float4*)(av + 4)  = *(const float4*)(xrow + kA + 4);
        *(float4*)(av + 8)  = *(const float4*)(xrow + kB);
        *(float4*)(av + 12) = *(const float4*)(xrow + kB + 4);
        *(float4*)(scv + 0)  = *(const float4*)(s_scale + kA);
        *(float4*)(scv + 4)  = *(const float4*)(s_scale + kA + 4);
        *(float4*)(scv + 8)  = *(const float4*)(s_scale + kB);
        *(float4*)(scv + 12) = *(const float4*)(s_scale + kB + 4);
        *(float4*)(shv + 0)  = *(const float4*)(s_shift + kA);
        *(float4*)(shv + 4)  = *(const float4*)(s_shift + kA + 4);
        *(float4*)(shv + 8)  = *(const float4*)(s_shift + kB);
        *(float4*)(shv + 12) = *(const float4*)(s_shift + kB + 4);

        // BN affine + ReLU + native bf16 convert (compiler picks cvt ops)
        v16bf afv;
        #pragma unroll
        for (int e = 0; e < 16; ++e) {
            float v = fmaf(av[e], scv[e], shv[e]);   // BN affine
            v = v > 0.0f ? v : 0.0f;                 // ReLU
            afv[e] = (__bf16)v;
        }

        // ---- B fragments + WMMA over 16 column tiles ----
        // B 32x16: lane n = mn, elems 0-7 -> K = k0 + half*8 + (0..7), elems 8-15 -> K = k0 + 16 + half*8 + (0..7).
        #pragma unroll
        for (int ct = 0; ct < 16; ++ct) {
            const unsigned short* wp =
                Wb + (size_t)(ct * 16 + mn) * D_DIM + k0 + half * 8;
            BFrag bf;
            bf.q[0] = *(const uint4*)(wp);
            bf.q[1] = *(const uint4*)(wp + 16);
            acc[ct] = __builtin_amdgcn_wmma_f32_16x16x32_bf16(
                false, afv, false, bf.v, (short)0, acc[ct], false, false);
        }
    }

    // ---- store C: VGPR r holds M = r + 8*half, N = mn ----
    const int rowBase = blockIdx.x * 128 + wave * 16 + 8 * half;
    #pragma unroll
    for (int ct = 0; ct < 16; ++ct) {
        const int col = ct * 16 + mn;
        const float bv = bias[col];
        #pragma unroll
        for (int r = 0; r < 8; ++r) {
            out[(size_t)(rowBase + r) * O_DIM + col] = acc[ct][r] + bv;
        }
    }
}

// ---------------------------------------------------------------------------
extern "C" void kernel_launch(void* const* d_in, const int* in_sizes, int n_in,
                              void* d_out, int out_size, void* d_ws, size_t ws_size,
                              hipStream_t stream) {
    const float* x     = (const float*)d_in[0];
    const float* gamma = (const float*)d_in[1];
    const float* beta  = (const float*)d_in[2];
    const float* W     = (const float*)d_in[3];
    const float* b     = (const float*)d_in[4];
    float* out = (float*)d_out;

    // workspace layout
    float* d_sum   = (float*)d_ws;                 // [4096]
    float* d_sumsq = d_sum + D_DIM;                // [4096]
    float* d_scale = d_sumsq + D_DIM;              // [4096]
    float* d_shift = d_scale + D_DIM;              // [4096]
    unsigned short* d_Wb = (unsigned short*)(d_shift + D_DIM); // [256*4096] bf16

    // zero the atomic accumulators (graph-capture safe)
    hipMemsetAsync(d_ws, 0, 2 * D_DIM * sizeof(float), stream);

    bn_stats_kernel<<<dim3(4, 128), 256, 0, stream>>>(x, d_sum, d_sumsq);
    bn_finalize_kernel<<<D_DIM / 256, 256, 0, stream>>>(d_sum, d_sumsq, gamma, beta,
                                                        d_scale, d_shift);
    wconv_kernel<<<(O_DIM * D_DIM) / (256 * 4), 256, 0, stream>>>(W, d_Wb);
    fused_bn_gemm_kernel<<<N_ROWS / 128, 256, 0, stream>>>(x, d_scale, d_shift,
                                                           d_Wb, b, out);
}
